// MultiheadAttention_16604343566884
// MI455X (gfx1250) — compile-verified, matrix operand order checked by reading, not run on hardware
//
#include <hip/hip_runtime.h>
#include <hip/hip_bf16.h>

// Problem constants (from reference)
#define BS   4
#define SEQ  2048
#define DM   1024
#define NH   16
#define DKH  64
// Mask fill: must be representable in f16 (row-max is reduced in packed f16).
// exp(-60000 - m) == 0 for any realistic m, and fully-masked rows still give
// the reference's uniform softmax instead of f16 -inf poisoning.
#define NEGV (-60000.0f)

typedef __attribute__((ext_vector_type(16))) _Float16 v16h;
typedef __attribute__((ext_vector_type(8)))  _Float16 v8h;
typedef __attribute__((ext_vector_type(2)))  _Float16 v2h;
typedef __attribute__((ext_vector_type(8)))  float    v8f;
typedef __attribute__((ext_vector_type(4)))  float    v4f;

// Assemble a 16-half WMMA fragment from two contiguous 8-half (16B) chunks.
// CDNA5 16-bit A/B packing: per lane, elements 0..7 are K=kb..kb+7,
// elements 8..15 are K=kb+16..kb+23, with kb = 8*(lane>>4) chosen by caller.
static __device__ __forceinline__ v16h frag_from_h(const _Float16* p0, const _Float16* p1) {
  v8h a = *(const v8h*)p0;
  v8h b = *(const v8h*)p1;
  v16h f;
#pragma unroll
  for (int i = 0; i < 8; ++i) { f[i] = a[i]; f[i + 8] = b[i]; }
  return f;
}

// Same, but source is fp32 converted in-flight to f16 (v_cvt_pk_f16_f32).
static __device__ __forceinline__ v16h frag_from_f(const float* p0, const float* p1) {
  v4f a0 = *(const v4f*)p0, a1 = *(const v4f*)(p0 + 4);
  v4f b0 = *(const v4f*)p1, b1 = *(const v4f*)(p1 + 4);
  v16h f;
#pragma unroll
  for (int i = 0; i < 4; ++i) {
    f[i]      = (_Float16)a0[i];
    f[i + 4]  = (_Float16)a1[i];
    f[i + 8]  = (_Float16)b0[i];
    f[i + 12] = (_Float16)b1[i];
  }
  return f;
}

static __device__ __forceinline__ v2h hmax2(v2h a, v2h b) {
  v2h r;
  r[0] = (a[0] > b[0]) ? a[0] : b[0];
  r[1] = (a[1] > b[1]) ? a[1] : b[1];
  return r;
}

__global__ void cvt_f32_to_f16(const float* __restrict__ src, _Float16* __restrict__ dst, int n) {
  int i = blockIdx.x * blockDim.x + threadIdx.x;
  if (i < n) dst[i] = (_Float16)src[i];
}

// out[m,n] = sum_k A[m,k] * W[n,k] + bias[n]   (i.e. X @ W.T + b)
// M = BS*SEQ = 8192, N = K = DM = 1024.
// Wave tile: 32x32 (2 A-frags x 2 B-frags -> 4 independent WMMA chains / k-step).
// Block: 128 threads = 4 waves in a 2x2 arrangement -> 64x64 block tile.
// MODE 0: A = fp32, out = f16 head-split   [b][h][s][dk]   (Q, K)
// MODE 1: A = fp32, out = f16 head-split-T [b][h][dk][s]   (V)
// MODE 2: A = f16,  out = fp32 flat [m][n]                  (final projection)
template <int MODE>
__global__ void proj_gemm(const float* __restrict__ Af,
                          const _Float16* __restrict__ Ah,
                          const _Float16* __restrict__ W16,
                          const float* __restrict__ bias,
                          void* __restrict__ out) {
  const int lane = threadIdx.x & 31;
  const int wave = threadIdx.x >> 5;
  const int m0 = blockIdx.y * 64 + (wave >> 1) * 32;
  const int n0 = blockIdx.x * 64 + (wave & 1) * 32;
  const int l15 = lane & 15;
  const int kb  = (lane >> 4) * 8;          // K sub-offset for this half-wave

  const _Float16* w0 = W16 + (size_t)(n0 + l15) * DM;        // B frag 0: cols n0..n0+15
  const _Float16* w1 = W16 + (size_t)(n0 + 16 + l15) * DM;   // B frag 1: cols +16

  v8f acc00 = {}, acc01 = {}, acc10 = {}, acc11 = {};

  for (int k0 = 0; k0 < DM; k0 += 32) {
    v16h a0, a1;
    if (MODE == 2) {
      const _Float16* ar0 = Ah + (size_t)(m0 + l15) * DM + k0;
      const _Float16* ar1 = Ah + (size_t)(m0 + 16 + l15) * DM + k0;
      a0 = frag_from_h(ar0 + kb, ar0 + kb + 16);
      a1 = frag_from_h(ar1 + kb, ar1 + kb + 16);
      __builtin_prefetch(ar0 + 128, 0, 3);
      __builtin_prefetch(ar1 + 128, 0, 3);
    } else {
      const float* ar0 = Af + (size_t)(m0 + l15) * DM + k0;
      const float* ar1 = Af + (size_t)(m0 + 16 + l15) * DM + k0;
      a0 = frag_from_f(ar0 + kb, ar0 + kb + 16);
      a1 = frag_from_f(ar1 + kb, ar1 + kb + 16);
      __builtin_prefetch(ar0 + 128, 0, 3);
      __builtin_prefetch(ar1 + 128, 0, 3);
    }
    const v16h b0 = frag_from_h(w0 + k0 + kb, w0 + k0 + kb + 16);
    const v16h b1 = frag_from_h(w1 + k0 + kb, w1 + k0 + kb + 16);
    __builtin_prefetch(w0 + k0 + 128, 0, 3);
    __builtin_prefetch(w1 + k0 + 128, 0, 3);
    acc00 = __builtin_amdgcn_wmma_f32_16x16x32_f16(false, a0, false, b0, (short)0, acc00, false, false);
    acc01 = __builtin_amdgcn_wmma_f32_16x16x32_f16(false, a0, false, b1, (short)0, acc01, false, false);
    acc10 = __builtin_amdgcn_wmma_f32_16x16x32_f16(false, a1, false, b0, (short)0, acc10, false, false);
    acc11 = __builtin_amdgcn_wmma_f32_16x16x32_f16(false, a1, false, b1, (short)0, acc11, false, false);
  }

  const int rowoff = (lane >> 4) * 8;
  const float bv0 = bias[n0 + l15];
  const float bv1 = bias[n0 + 16 + l15];
#pragma unroll
  for (int mt = 0; mt < 2; ++mt) {
#pragma unroll
    for (int nt = 0; nt < 2; ++nt) {
      const v8f acc = (mt == 0) ? (nt == 0 ? acc00 : acc01)
                                : (nt == 0 ? acc10 : acc11);
      const float bval = nt == 0 ? bv0 : bv1;
      const int n = n0 + nt * 16 + l15;     // C layout: col = lane & 15
#pragma unroll
      for (int r = 0; r < 8; ++r) {
        const int m = m0 + mt * 16 + r + rowoff;  // row = r + 8*(lane>=16)
        const float v = acc[r] + bval;
        if (MODE == 2) {
          ((float*)out)[(size_t)m * DM + n] = v;
        } else {
          const int b_ = m >> 11, s = m & (SEQ - 1);
          const int h  = n >> 6,  d = n & (DKH - 1);
          if (MODE == 0)
            ((_Float16*)out)[((size_t)(b_ * NH + h) * SEQ + s) * DKH + d] = (_Float16)v;
          else
            ((_Float16*)out)[((size_t)(b_ * NH + h) * DKH + d) * SEQ + s] = (_Float16)v;
        }
      }
    }
  }
}

// Flash attention: one wave per 16-row query tile, streaming keys 32 at a time.
// S-tile: A = Q (16x32 over dk, 2 frags), B = K rows (contiguous per lane).
// PV:     A = P (rebuilt from LDS), B = V^T rows (contiguous per lane).
// Softmax denominator comes from a P x ones WMMA (no shuffle row-sums);
// row-max is reduced in packed f16 (any consistent m cancels exactly).
__global__ void flash_attn(const _Float16* __restrict__ Qh,
                           const _Float16* __restrict__ Kh,
                           const _Float16* __restrict__ Vt,
                           const unsigned char* __restrict__ mask,
                           _Float16* __restrict__ attn_out) {
  __shared__ __align__(16) _Float16 psh[4][16][32];   // per-wave 16x32 P tile

  const int lane = threadIdx.x & 31;
  const int wave = threadIdx.x >> 5;
  const int bh = blockIdx.y;
  const int b_ = bh >> 4, h = bh & 15;
  const int q0 = (blockIdx.x * 4 + wave) * 16;
  const int l15 = lane & 15;
  const int kb  = (lane >> 4) * 8;
  const int rowoff = (lane >> 4) * 8;
  const float scale = 0.125f;               // 1/sqrt(64)

  const _Float16* Qb = Qh + (size_t)bh * SEQ * DKH;
  const _Float16* Kb = Kh + (size_t)bh * SEQ * DKH;
  const _Float16* Vb = Vt + (size_t)bh * DKH * SEQ;
  const unsigned char* mb = mask + (size_t)b_ * SEQ;

  // Q fragments (held for whole loop): A row = q0 + (lane&15), dk 0..31 / 32..63
  const _Float16* qrow = Qb + (size_t)(q0 + l15) * DKH;
  const v16h aq0 = frag_from_h(qrow + kb, qrow + kb + 16);
  const v16h aq1 = frag_from_h(qrow + 32 + kb, qrow + 32 + kb + 16);

  v16h bones;                                // all-ones B fragment for l = P @ 1
#pragma unroll
  for (int i = 0; i < 16; ++i) bones[i] = (_Float16)1.0f;

  v8f acc0 = {}, acc1 = {}, acc2 = {}, acc3 = {}, accL = {};
  float mrun[8];
#pragma unroll
  for (int r = 0; r < 8; ++r) mrun[r] = -1e30f;

  for (int j0 = 0; j0 < SEQ; j0 += 32) {
    // ---- S = Q K^T, two 16-col tiles ----
    const _Float16* kr0 = Kb + (size_t)(j0 + l15) * DKH;
    const _Float16* kr1 = Kb + (size_t)(j0 + 16 + l15) * DKH;
    v8f s0 = {}, s1 = {};
    v16h bk;
    bk = frag_from_h(kr0 + kb, kr0 + kb + 16);
    s0 = __builtin_amdgcn_wmma_f32_16x16x32_f16(false, aq0, false, bk, (short)0, s0, false, false);
    bk = frag_from_h(kr0 + 32 + kb, kr0 + 32 + kb + 16);
    s0 = __builtin_amdgcn_wmma_f32_16x16x32_f16(false, aq1, false, bk, (short)0, s0, false, false);
    bk = frag_from_h(kr1 + kb, kr1 + kb + 16);
    s1 = __builtin_amdgcn_wmma_f32_16x16x32_f16(false, aq0, false, bk, (short)0, s1, false, false);
    bk = frag_from_h(kr1 + 32 + kb, kr1 + 32 + kb + 16);
    s1 = __builtin_amdgcn_wmma_f32_16x16x32_f16(false, aq1, false, bk, (short)0, s1, false, false);

    const bool msk0 = mb[j0 + l15] != 0;         // column mask, one col per lane
    const bool msk1 = mb[j0 + 16 + l15] != 0;

    // ---- scale + mask, local row max ----
    float v0a[8], v1a[8];
    v2h pm[4];
#pragma unroll
    for (int r = 0; r < 8; ++r) {
      const float v0 = msk0 ? NEGV : s0[r] * scale;
      const float v1 = msk1 ? NEGV : s1[r] * scale;
      v0a[r] = v0; v1a[r] = v1;
      const float mx = fmaxf(v0, v1);
      if (r & 1) pm[r >> 1][1] = (_Float16)mx;
      else       pm[r >> 1][0] = (_Float16)mx;
    }
    // ---- cross-lane max: 4 packed dwords, 4 xor steps within 16-lane halves ----
#pragma unroll
    for (int off = 8; off >= 1; off >>= 1) {
#pragma unroll
      for (int k = 0; k < 4; ++k) {
        int u = __builtin_bit_cast(int, pm[k]);
        u = __shfl_xor(u, off, 16);
        pm[k] = hmax2(pm[k], __builtin_bit_cast(v2h, u));
      }
    }

    // ---- online softmax update; P -> LDS; rescale accumulators ----
#pragma unroll
    for (int r = 0; r < 8; ++r) {
      const float rowmax = (float)((r & 1) ? pm[r >> 1][1] : pm[r >> 1][0]);
      const float mnew  = fmaxf(mrun[r], rowmax);
      const float alpha = __expf(mrun[r] - mnew);
      const float p0 = __expf(v0a[r] - mnew);
      const float p1 = __expf(v1a[r] - mnew);
      mrun[r] = mnew;
      acc0[r] *= alpha; acc1[r] *= alpha; acc2[r] *= alpha; acc3[r] *= alpha;
      accL[r] *= alpha;
      // C-layout -> LDS (row = r + 8*(lane>=16), cols lane&15 / +16)
      psh[wave][r + rowoff][l15]      = (_Float16)p0;
      psh[wave][r + rowoff][l15 + 16] = (_Float16)p1;
    }
    asm volatile("s_wait_dscnt 0x0" ::: "memory");  // same-wave DS ordering

    // ---- rebuild P as an A fragment (lane row = lane&15, K pattern via kb) ----
    const _Float16* prow = &psh[wave][l15][0];
    const v16h ap = frag_from_h(prow + kb, prow + kb + 16);

    // ---- l accumulation via matrix pipe: accL += P @ ones ----
    accL = __builtin_amdgcn_wmma_f32_16x16x32_f16(false, ap, false, bones, (short)0, accL, false, false);

    // ---- PV: 4 dk tiles, V^T rows contiguous in j per lane ----
    const _Float16* vr0 = Vb + (size_t)(0 * 16 + l15) * SEQ + j0;
    const _Float16* vr1 = Vb + (size_t)(1 * 16 + l15) * SEQ + j0;
    const _Float16* vr2 = Vb + (size_t)(2 * 16 + l15) * SEQ + j0;
    const _Float16* vr3 = Vb + (size_t)(3 * 16 + l15) * SEQ + j0;
    v16h bv;
    bv = frag_from_h(vr0 + kb, vr0 + kb + 16);
    acc0 = __builtin_amdgcn_wmma_f32_16x16x32_f16(false, ap, false, bv, (short)0, acc0, false, false);
    bv = frag_from_h(vr1 + kb, vr1 + kb + 16);
    acc1 = __builtin_amdgcn_wmma_f32_16x16x32_f16(false, ap, false, bv, (short)0, acc1, false, false);
    bv = frag_from_h(vr2 + kb, vr2 + kb + 16);
    acc2 = __builtin_amdgcn_wmma_f32_16x16x32_f16(false, ap, false, bv, (short)0, acc2, false, false);
    bv = frag_from_h(vr3 + kb, vr3 + kb + 16);
    acc3 = __builtin_amdgcn_wmma_f32_16x16x32_f16(false, ap, false, bv, (short)0, acc3, false, false);
  }

  // ---- normalize (l lives in accL, same value in every column) and store ----
#pragma unroll
  for (int r = 0; r < 8; ++r) {
    const int q = q0 + r + rowoff;
    const float invl = 1.0f / accL[r];
    _Float16* orow = attn_out + ((size_t)(b_ * SEQ + q)) * DM + h * DKH;
    orow[0 * 16 + l15] = (_Float16)(acc0[r] * invl);
    orow[1 * 16 + l15] = (_Float16)(acc1[r] * invl);
    orow[2 * 16 + l15] = (_Float16)(acc2[r] * invl);
    orow[3 * 16 + l15] = (_Float16)(acc3[r] * invl);
  }
}

extern "C" void kernel_launch(void* const* d_in, const int* in_sizes, int n_in,
                              void* d_out, int out_size, void* d_ws, size_t ws_size,
                              hipStream_t stream) {
  (void)in_sizes; (void)n_in; (void)out_size; (void)ws_size;
  const float* query = (const float*)d_in[0];
  const float* key   = (const float*)d_in[1];
  const float* value = (const float*)d_in[2];
  const unsigned char* mask = (const unsigned char*)d_in[3];
  const float* Wq = (const float*)d_in[4];
  const float* bq = (const float*)d_in[5];
  const float* Wk = (const float*)d_in[6];
  const float* bk = (const float*)d_in[7];
  const float* Wv = (const float*)d_in[8];
  const float* bv = (const float*)d_in[9];
  const float* Wo = (const float*)d_in[10];
  const float* bo = (const float*)d_in[11];
  float* out = (float*)d_out;

  // Workspace layout (f16): 4 weights (2MB each) + Q + K + V^T + attn (16MB each) = ~76MB
  char* ws = (char*)d_ws;
  const size_t WB = (size_t)DM * DM * sizeof(_Float16);
  const size_t QB = (size_t)BS * NH * SEQ * DKH * sizeof(_Float16);
  _Float16* W16q = (_Float16*)(ws);
  _Float16* W16k = (_Float16*)(ws + WB);
  _Float16* W16v = (_Float16*)(ws + 2 * WB);
  _Float16* W16o = (_Float16*)(ws + 3 * WB);
  _Float16* Qh   = (_Float16*)(ws + 4 * WB);
  _Float16* Kh   = (_Float16*)(ws + 4 * WB + QB);
  _Float16* Vt   = (_Float16*)(ws + 4 * WB + 2 * QB);
  _Float16* A16  = (_Float16*)(ws + 4 * WB + 3 * QB);

  const int wn = DM * DM;
  dim3 cvtg((wn + 255) / 256);
  cvt_f32_to_f16<<<cvtg, 256, 0, stream>>>(Wq, W16q, wn);
  cvt_f32_to_f16<<<cvtg, 256, 0, stream>>>(Wk, W16k, wn);
  cvt_f32_to_f16<<<cvtg, 256, 0, stream>>>(Wv, W16v, wn);
  cvt_f32_to_f16<<<cvtg, 256, 0, stream>>>(Wo, W16o, wn);

  dim3 gg(DM / 64, (BS * SEQ) / 64);         // 16 x 128, 4 waves/block (2x2 of 32x32)
  proj_gemm<0><<<gg, 128, 0, stream>>>(query, nullptr, W16q, bq, Qh);
  proj_gemm<0><<<gg, 128, 0, stream>>>(key,   nullptr, W16k, bk, Kh);
  proj_gemm<1><<<gg, 128, 0, stream>>>(value, nullptr, W16v, bv, Vt);

  dim3 ag(SEQ / (16 * 4), BS * NH);          // 32 x 64, 4 waves/block
  flash_attn<<<ag, 128, 0, stream>>>(Qh, Kh, Vt, mask, A16);

  proj_gemm<2><<<gg, 128, 0, stream>>>(nullptr, A16, W16o, bo, out);
}
